// DRAW_51247549776040
// MI455X (gfx1250) — matrix operand order repaired and checked by reading, not compiled, not run on hardware
//
#include <hip/hip_runtime.h>
#include <hip/hip_bf16.h>

typedef __bf16 bf16_t;
typedef __attribute__((ext_vector_type(16))) __bf16 v16bf;
typedef __attribute__((ext_vector_type(8)))  float  v8f;

// ---------------------------------------------------------------- helpers ---
static __device__ __forceinline__ v8f wmma_bf16(v16bf a, v16bf b, v8f c) {
  return __builtin_amdgcn_wmma_f32_16x16x32_bf16(false, a, false, b, (short)0, c,
                                                 false, false);
}

// two 16B chunks -> one v16bf fragment (A-side: K [kb,kb+8) and [kb+16,kb+24))
static __device__ __forceinline__ v16bf ld2x16(const bf16_t* p0,
                                               const bf16_t* p1) {
  v16bf r;
  uint4* d = reinterpret_cast<uint4*>(&r);
  d[0] = *reinterpret_cast<const uint4*>(p0);
  d[1] = *reinterpret_cast<const uint4*>(p1);
  return r;
}
// 32B contiguous -> v16bf (packed B fragments)
static __device__ __forceinline__ v16bf ld32(const bf16_t* p) {
  return ld2x16(p, p + 8);
}

static __device__ __forceinline__ float sigm(float x) {
  return 1.0f / (1.0f + expf(-x));
}

// Fragment-native packed B layout for a K x N matrix (NT = N/16):
// element (k,n) lives at ((k/32)*NT + n/16)*512 + (((k/8)&1)*16 + (n&15))*16
//                         + ((k/16)&1)*8 + (k&7)
// so lane L of tile (kt,nt) reads 16 contiguous bf16 (= 2 x b128).
static __device__ __forceinline__ int pack_idx(int k, int n, int NT) {
  return (((k >> 5) * NT + (n >> 4)) * 32 + (((k >> 3) & 1) << 4) + (n & 15)) *
             16 +
         (((k >> 4) & 1) << 3) + (k & 7);
}

// ------------------------------------------------------------- main GEMM ---
// C[M,N] = A[M,K] * Bpacked[K,N] (+bias[N]); A bf16 row-major, C fp32.
// block tile 128x64, 8 waves, wave tile 32x32 (2x2 WMMA), K-step 32.
// A tiles staged memory->LDS with GLOBAL_LOAD_ASYNC_TO_LDS_B128 (ASYNCcnt),
// double-buffered so the DMA of tile k+1 overlaps the WMMAs of tile k.
#define TM 128
#define TN 64
#define TK 32

__global__ __launch_bounds__(256) void gemm_bf16_wmma(
    const bf16_t* __restrict__ A, const bf16_t* __restrict__ Bp,
    float* __restrict__ C, const float* __restrict__ bias, int M, int N,
    int K) {
  __shared__ bf16_t As[2][TM][TK + 8];  // pitch 40 elems = 80B (16B aligned)

  const int tid = threadIdx.x;
  const int wave = tid >> 5, lane = tid & 31;
  const int hl = lane >> 4, l16 = lane & 15;
  const int wM = wave >> 1, wN = wave & 1;
  const int tileM = blockIdx.y * TM;
  const int tileN = blockIdx.x * TN;
  const int NT = N >> 4;

  // per-thread staging geometry: 512 16B chunks per tile, 2 per thread
  const int sr0 = tid >> 2, scc = (tid & 3) * 8;
  const bf16_t* ag0 = A + (size_t)(tileM + sr0) * K + scc;
  const bf16_t* ag1 = A + (size_t)(tileM + sr0 + 64) * K + scc;

  v8f acc[2][2] = {};

#define STAGE_A(buf, k0)                                                       \
  do {                                                                         \
    unsigned l0 = (unsigned)(uintptr_t)&As[buf][sr0][scc];                     \
    unsigned l1 = (unsigned)(uintptr_t)&As[buf][sr0 + 64][scc];                \
    unsigned long long g0 = (unsigned long long)(uintptr_t)(ag0 + (k0));       \
    unsigned long long g1 = (unsigned long long)(uintptr_t)(ag1 + (k0));       \
    asm volatile("global_load_async_to_lds_b128 %0, %1, off" ::"v"(l0),        \
                 "v"(g0)                                                       \
                 : "memory");                                                  \
    asm volatile("global_load_async_to_lds_b128 %0, %1, off" ::"v"(l1),        \
                 "v"(g1)                                                       \
                 : "memory");                                                  \
  } while (0)

  STAGE_A(0, 0);
  int s = 0;
  for (int k0 = 0; k0 < K; k0 += TK, s ^= 1) {
    asm volatile("s_wait_asynccnt 0x0" ::: "memory");
    __syncthreads();  // publish As[s]; all waves done reading As[s^1]
    if (k0 + TK < K) {
      STAGE_A(s ^ 1, k0 + TK);  // DMA next tile under this tile's compute
      __builtin_prefetch(
          Bp + (((size_t)((k0 + TK) >> 5) * NT + (tileN >> 4)) * 32 + lane) * 16,
          0, 1);
    }

    v16bf af[2], bfr[2];
#pragma unroll
    for (int mi = 0; mi < 2; ++mi) {
      int row = wM * 32 + mi * 16 + l16, kb = hl * 8;
      af[mi] = ld2x16(&As[s][row][kb], &As[s][row][kb + 16]);
    }
#pragma unroll
    for (int ni = 0; ni < 2; ++ni) {
      int nt = (tileN >> 4) + wN * 2 + ni;
      bfr[ni] = ld32(Bp + (((size_t)(k0 >> 5) * NT + nt) * 32 + lane) * 16);
    }
#pragma unroll
    for (int mi = 0; mi < 2; ++mi)
#pragma unroll
      for (int ni = 0; ni < 2; ++ni)
        acc[mi][ni] = wmma_bf16(af[mi], bfr[ni], acc[mi][ni]);
  }
#undef STAGE_A

#pragma unroll
  for (int mi = 0; mi < 2; ++mi)
#pragma unroll
    for (int ni = 0; ni < 2; ++ni)
#pragma unroll
      for (int r = 0; r < 8; ++r) {
        int gm = tileM + wM * 32 + mi * 16 + r + 8 * hl;
        int gn = tileN + wN * 32 + ni * 16 + l16;
        float v = acc[mi][ni][r];
        if (bias) v += bias[gn];
        C[(size_t)gm * N + gn] = v;
      }
}

// ----------------------------------------------------- batched read attn ---
// per batch b: for img in {x, x_hat} (both B-packed):
//   tmp = Fy(32x128) @ img(128x128); out = tmp @ FxT(128x32); enc_in = g*out
__global__ __launch_bounds__(256) void read_attn_kernel(
    const bf16_t* __restrict__ Fy, const bf16_t* __restrict__ FxTp,
    const bf16_t* __restrict__ imgp, const bf16_t* __restrict__ xhatp,
    const float* __restrict__ gamma, bf16_t* __restrict__ enc_in) {
  __shared__ bf16_t sT[32][136];  // tmp (32x128) bf16, pitch 272B
  const int b = blockIdx.x;
  const bf16_t* Fyb = Fy + (size_t)b * 4096;
  const bf16_t* FxTb = FxTp + (size_t)b * 4096;
  const float g = gamma[b];
  const int tid = threadIdx.x, wave = tid >> 5, lane = tid & 31;
  const int hl = lane >> 4, l16 = lane & 15;

  for (int im = 0; im < 2; ++im) {
    const bf16_t* Mb = (im == 0 ? imgp : xhatp) + (size_t)b * 16384;
    if (im) __syncthreads();
    // GEMM1: wave w covers N cols [16w,16w+16); img packed with NT=8
    v8f acc[2] = {};
    for (int k0 = 0; k0 < 128; k0 += 32) {
      v16bf bfr = ld32(Mb + ((size_t)(k0 >> 5) * 8 + wave) * 512 + lane * 16);
#pragma unroll
      for (int mi = 0; mi < 2; ++mi) {
        const bf16_t* p0 = Fyb + (mi * 16 + l16) * 128 + k0 + hl * 8;
        acc[mi] = wmma_bf16(ld2x16(p0, p0 + 16), bfr, acc[mi]);
      }
    }
#pragma unroll
    for (int mi = 0; mi < 2; ++mi)
#pragma unroll
      for (int r = 0; r < 8; ++r)
        sT[mi * 16 + r + 8 * hl][wave * 16 + l16] = (bf16_t)acc[mi][r];
    __syncthreads();
    // GEMM2: waves 0..3 each one 16x16 tile of 32x32 out; FxT packed NT=2
    if (wave < 4) {
      int mi = wave >> 1, ni = wave & 1;
      v8f o = {};
      for (int k0 = 0; k0 < 128; k0 += 32) {
        const bf16_t* p0 = &sT[mi * 16 + l16][k0 + hl * 8];
        v16bf bfr =
            ld32(FxTb + ((size_t)(k0 >> 5) * 2 + ni) * 512 + lane * 16);
        o = wmma_bf16(ld2x16(p0, p0 + 16), bfr, o);
      }
      bf16_t* dst = enc_in + (size_t)b * 2560 + im * 1024;
#pragma unroll
      for (int r = 0; r < 8; ++r) {
        int gm = mi * 16 + r + 8 * hl, gn = ni * 16 + l16;
        dst[gm * 32 + gn] = (bf16_t)(g * o[r]);
      }
    }
  }
}

// ---------------------------------------------------- batched write attn ---
// per batch b: tmp = FyT(128x32) @ w(32x32); reconsts += (tmp @ Fxw(32x128))/g
__global__ __launch_bounds__(256) void write_attn_kernel(
    const bf16_t* __restrict__ FyT, const bf16_t* __restrict__ Fxwp,
    const bf16_t* __restrict__ wbp, const float* __restrict__ gamma,
    float* __restrict__ reconsts) {
  __shared__ bf16_t sT[128][40];  // tmp (128x32) bf16
  const int b = blockIdx.x;
  const bf16_t* FyTb = FyT + (size_t)b * 4096;
  const bf16_t* Fxwb = Fxwp + (size_t)b * 4096;
  const bf16_t* wbb = wbp + (size_t)b * 1024;
  const float ginv = 1.0f / gamma[b];
  const int tid = threadIdx.x, wave = tid >> 5, lane = tid & 31;
  const int hl = lane >> 4, l16 = lane & 15;

  {  // GEMM1: wave w -> M tile w; ni in {0,1}; w packed NT=2; K=32 one iter
    const bf16_t* p0 = FyTb + (wave * 16 + l16) * 32 + hl * 8;
    v16bf af = ld2x16(p0, p0 + 16);
#pragma unroll
    for (int ni = 0; ni < 2; ++ni) {
      v16bf bfr = ld32(wbb + (size_t)ni * 512 + lane * 16);
      v8f o = {};
      o = wmma_bf16(af, bfr, o);
#pragma unroll
      for (int r = 0; r < 8; ++r)
        sT[wave * 16 + r + 8 * hl][ni * 16 + l16] = (bf16_t)o[r];
    }
  }
  __syncthreads();
  {  // GEMM2: wave w -> M tile w, 8 N tiles; Fxw packed NT=8; K=32 one iter
    const bf16_t* p0 = &sT[wave * 16 + l16][hl * 8];
    v16bf af = ld2x16(p0, p0 + 16);
    float* Rb = reconsts + (size_t)b * 16384;
#pragma unroll
    for (int ni = 0; ni < 8; ++ni) {
      v16bf bfr = ld32(Fxwb + (size_t)ni * 512 + lane * 16);
      v8f o = {};
      o = wmma_bf16(af, bfr, o);
#pragma unroll
      for (int r = 0; r < 8; ++r) {
        int gm = wave * 16 + r + 8 * hl, gn = ni * 16 + l16;
        Rb[gm * 128 + gn] += ginv * o[r];
      }
    }
  }
}

// ------------------------------------------------- attention filterbanks ---
// p = h_dec @ Wa^T + ba; normalized filterbanks:
//  Fx emitted B-packed twice: as FxT (K=pixel,N=grid, NT=2) for the read path
//  and as Fxw (K=grid,N=pixel, NT=8) for the write path.
//  Fy emitted row-major [32,128] (read A-side) and transposed [128,32]
//  (write A-side).
__global__ __launch_bounds__(128) void fb_kernel(
    const float* __restrict__ hdec, const float* __restrict__ Wa,
    const float* __restrict__ ba, float* __restrict__ gamma_out,
    bf16_t* __restrict__ Fp1, bf16_t* __restrict__ Fp2,
    bf16_t* __restrict__ Fy_n, bf16_t* __restrict__ Fy_t) {
  __shared__ float red[5][128];
  __shared__ float vals[32][128];
  __shared__ float rsum[32][4];
  __shared__ float sc[4];
  const int b = blockIdx.x, t = threadIdx.x;
  const float* h = hdec + (size_t)b * 512;
  float p[5] = {0, 0, 0, 0, 0};
  for (int k = t; k < 512; k += 128) {
    float hv = h[k];
#pragma unroll
    for (int j = 0; j < 5; ++j) p[j] += hv * Wa[j * 512 + k];
  }
#pragma unroll
  for (int j = 0; j < 5; ++j) red[j][t] = p[j];
  __syncthreads();
  for (int off = 64; off > 0; off >>= 1) {
    if (t < off)
#pragma unroll
      for (int j = 0; j < 5; ++j) red[j][t] += red[j][t + off];
    __syncthreads();
  }
  if (t == 0) {
    float q0 = red[0][0] + ba[0], q1 = red[1][0] + ba[1];
    float q2 = red[2][0] + ba[2], q3 = red[3][0] + ba[3];
    float q4 = red[4][0] + ba[4];
    sc[0] = 0.5f * 129.0f * (q0 + 1.0f);  // g_x
    sc[1] = 0.5f * 129.0f * (q1 + 1.0f);  // g_y
    sc[2] = 2.0f * expf(q2);              // two_sigma_sq
    sc[3] = 127.0f * expf(q3) / 31.0f;    // delta
    gamma_out[b] = expf(q4);
  }
  __syncthreads();
  const int i = t >> 2, qd = t & 3;  // grid row 0..31, pixel quarter
  for (int which = 0; which < 2; ++which) {
    float mu = sc[which] + ((float)i - 15.5f) * sc[3];
    float tss = sc[2], part = 0.0f;
    for (int a0 = 0; a0 < 32; ++a0) {
      int a = qd * 32 + a0;
      float d = (float)a - mu;
      float v = expf(-d * d / tss);
      vals[i][a] = v;
      part += v;
    }
    rsum[i][qd] = part;
    __syncthreads();
    if (qd == 0) {
      float s = rsum[i][0] + rsum[i][1] + rsum[i][2] + rsum[i][3];
      rsum[i][0] = 1.0f / fmaxf(s, 1e-8f);
    }
    __syncthreads();
    float rinv = rsum[i][0];
    if (which == 0) {
      bf16_t* P1 = Fp1 + (size_t)b * 4096;
      bf16_t* P2 = Fp2 + (size_t)b * 4096;
      for (int a0 = 0; a0 < 32; ++a0) {
        int a = qd * 32 + a0;
        float nv = vals[i][a] * rinv;
        P1[pack_idx(a, i, 2)] = (bf16_t)nv;  // FxT packed (read)
        P2[pack_idx(i, a, 8)] = (bf16_t)nv;  // Fxw packed (write)
      }
    } else {
      bf16_t* Fn = Fy_n + (size_t)b * 4096;
      bf16_t* Ft = Fy_t + (size_t)b * 4096;
      for (int a0 = 0; a0 < 32; ++a0) {
        int a = qd * 32 + a0;
        float nv = vals[i][a] * rinv;
        Fn[i * 128 + a] = (bf16_t)nv;  // Fy row-major (read A)
        Ft[a * 32 + i] = (bf16_t)nv;   // FyT row-major (write A)
      }
    }
    __syncthreads();
  }
}

// ---------------------------------------------------------- fused pieces ---
__global__ void gru_fuse_kernel(const float* __restrict__ gi,
                                const float* __restrict__ gh,
                                float* __restrict__ h, bf16_t* __restrict__ hbf,
                                bf16_t* __restrict__ tail) {
  int idx = blockIdx.x * blockDim.x + threadIdx.x;
  if (idx >= 256 * 512) return;
  int b = idx >> 9, j = idx & 511;
  size_t base = (size_t)b * 1536;
  float r = sigm(gi[base + j] + gh[base + j]);
  float z = sigm(gi[base + 512 + j] + gh[base + 512 + j]);
  float n = tanhf(gi[base + 1024 + j] + r * gh[base + 1024 + j]);
  float hv = (1.0f - z) * n + z * h[idx];
  h[idx] = hv;
  hbf[idx] = (bf16_t)hv;
  if (tail) tail[(size_t)b * 2560 + 2048 + j] = (bf16_t)hv;
}

__global__ void zfuse_kernel(const float* __restrict__ gml,
                             const float* __restrict__ eps_t,
                             bf16_t* __restrict__ zb) {
  int idx = blockIdx.x * blockDim.x + threadIdx.x;
  if (idx >= 256 * 128) return;
  int b = idx >> 7, j = idx & 127;
  float mu = gml[b * 256 + j];
  float ls = gml[b * 256 + 128 + j];
  zb[idx] = (bf16_t)(mu + eps_t[idx] * expf(ls));
}

// x_hat = x - sigmoid(reconsts[last batch row]) (faithful broadcast quirk),
// emitted directly in B-packed (K=row,N=col, NT=8) layout per batch image.
__global__ void xhat_pack_kernel(const float* __restrict__ x,
                                 const float* __restrict__ rec,
                                 bf16_t* __restrict__ out) {
  int idx = blockIdx.x * blockDim.x + threadIdx.x;
  if (idx >= 256 * 16384) return;
  int b = idx >> 14, i = idx & 16383;
  float v = x[idx] - sigm(rec[255 * 16384 + i]);
  out[(size_t)b * 16384 + pack_idx(i >> 7, i & 127, 8)] = (bf16_t)v;
}

__global__ void img_pack_kernel(const float* __restrict__ x,
                                bf16_t* __restrict__ out) {
  int idx = blockIdx.x * blockDim.x + threadIdx.x;
  if (idx >= 256 * 16384) return;
  int b = idx >> 14, i = idx & 16383;
  out[(size_t)b * 16384 + pack_idx(i >> 7, i & 127, 8)] = (bf16_t)x[idx];
}

// repack write patch w [256,32,32] (f32 rows of the wr-GEMM C) into B-packed
__global__ void wpack_kernel(const float* __restrict__ wlin,
                             bf16_t* __restrict__ wp) {
  int idx = blockIdx.x * blockDim.x + threadIdx.x;
  if (idx >= 256 * 1024) return;
  int b = idx >> 10, c = idx & 1023;
  int k = c >> 5, n = c & 31;
  wp[(size_t)b * 1024 + pack_idx(k, n, 2)] = (bf16_t)wlin[idx];
}

__global__ void sigmoid_out_kernel(const float* __restrict__ rec,
                                   float* __restrict__ out, int n) {
  int idx = blockIdx.x * blockDim.x + threadIdx.x;
  if (idx < n) out[idx] = sigm(rec[idx]);
}

// ------------------------------------------------------------ setup bits ---
// pack weight W[Nsrc,K] (fp32, used as B = W^T) into fragment-native bf16
__global__ void pack_w_kernel(const float* __restrict__ W,
                              bf16_t* __restrict__ Bp, int K, int NT, int nOff,
                              int total) {
  int idx = blockIdx.x * blockDim.x + threadIdx.x;
  if (idx >= total) return;
  int n = idx / K + nOff, k = idx - (idx / K) * K;
  Bp[pack_idx(k, n, NT)] = (bf16_t)W[idx];
}

__global__ void bias_concat_kernel(const float* __restrict__ bmu,
                                   const float* __restrict__ bls,
                                   float* __restrict__ bml) {
  int t = threadIdx.x;
  if (t < 256) bml[t] = (t < 128) ? bmu[t] : bls[t - 128];
}

__global__ void zero_f32_kernel(float* p, int n) {
  int idx = blockIdx.x * blockDim.x + threadIdx.x;
  if (idx < n) p[idx] = 0.0f;
}
__global__ void zero_bf16_kernel(bf16_t* p, int n) {
  int idx = blockIdx.x * blockDim.x + threadIdx.x;
  if (idx < n) p[idx] = (bf16_t)0.0f;
}

// ---------------------------------------------------------------- driver ---
extern "C" void kernel_launch(void* const* d_in, const int* in_sizes, int n_in,
                              void* d_out, int out_size, void* d_ws,
                              size_t ws_size, hipStream_t stream) {
  const float* x = (const float*)d_in[0];
  const float* eps = (const float*)d_in[1];
  const float* W_ih_enc = (const float*)d_in[2];
  const float* W_hh_enc = (const float*)d_in[3];
  const float* b_ih_enc = (const float*)d_in[4];
  const float* b_hh_enc = (const float*)d_in[5];
  const float* W_ih_dec = (const float*)d_in[6];
  const float* W_hh_dec = (const float*)d_in[7];
  const float* b_ih_dec = (const float*)d_in[8];
  const float* b_hh_dec = (const float*)d_in[9];
  const float* W_mu = (const float*)d_in[10];
  const float* b_mu = (const float*)d_in[11];
  const float* W_ls = (const float*)d_in[12];
  const float* b_ls = (const float*)d_in[13];
  const float* W_wr = (const float*)d_in[14];
  const float* b_wr = (const float*)d_in[15];
  const float* W_ra = (const float*)d_in[16];
  const float* b_ra = (const float*)d_in[17];
  const float* W_wa = (const float*)d_in[18];
  const float* b_wa = (const float*)d_in[19];

  char* ws = (char*)d_ws;
  size_t cur = 0;
  auto alloc = [&](size_t bytes) -> char* {
    char* p = ws + cur;
    cur = (cur + bytes + 255) & ~(size_t)255;
    return p;
  };
  float* reconsts = (float*)alloc(4194304ull * 4);
  bf16_t* img_p = (bf16_t*)alloc(4194304ull * 2);
  bf16_t* xhat_p = (bf16_t*)alloc(4194304ull * 2);
  bf16_t* WpIE = (bf16_t*)alloc(2560ull * 1536 * 2);
  bf16_t* WpHE = (bf16_t*)alloc(512ull * 1536 * 2);
  bf16_t* WpID = (bf16_t*)alloc(128ull * 1536 * 2);
  bf16_t* WpHD = (bf16_t*)alloc(512ull * 1536 * 2);
  bf16_t* WpML = (bf16_t*)alloc(512ull * 256 * 2);
  bf16_t* WpWR = (bf16_t*)alloc(512ull * 1024 * 2);
  float* bml = (float*)alloc(256 * 4);
  float* h_enc = (float*)alloc(131072ull * 4);
  float* h_dec = (float*)alloc(131072ull * 4);
  bf16_t* h_enc_bf = (bf16_t*)alloc(131072ull * 2);
  bf16_t* h_dec_bf = (bf16_t*)alloc(131072ull * 2);
  bf16_t* enc_in = (bf16_t*)alloc(655360ull * 2);
  float* gi = (float*)alloc(393216ull * 4);
  float* gh = (float*)alloc(393216ull * 4);
  float* gml = (float*)alloc(65536ull * 4);
  bf16_t* zb = (bf16_t*)alloc(32768ull * 2);
  bf16_t* wbp = (bf16_t*)alloc(262144ull * 2);
  bf16_t* Fx_p1 = (bf16_t*)alloc(1048576ull * 2);  // FxT packed (read)
  bf16_t* Fx_p2 = (bf16_t*)alloc(1048576ull * 2);  // Fxw packed (write)
  bf16_t* Fy_n = (bf16_t*)alloc(1048576ull * 2);
  bf16_t* Fy_t = (bf16_t*)alloc(1048576ull * 2);
  float* gamma_r = (float*)alloc(256 * 4);
  float* gamma_w = (float*)alloc(256 * 4);

  auto grd = [](int n) { return (n + 255) / 256; };
  // weights -> fragment-native packed bf16 (B = W^T)
  pack_w_kernel<<<grd(1536 * 2560), 256, 0, stream>>>(W_ih_enc, WpIE, 2560, 96,
                                                      0, 1536 * 2560);
  pack_w_kernel<<<grd(1536 * 512), 256, 0, stream>>>(W_hh_enc, WpHE, 512, 96, 0,
                                                     1536 * 512);
  pack_w_kernel<<<grd(1536 * 128), 256, 0, stream>>>(W_ih_dec, WpID, 128, 96, 0,
                                                     1536 * 128);
  pack_w_kernel<<<grd(1536 * 512), 256, 0, stream>>>(W_hh_dec, WpHD, 512, 96, 0,
                                                     1536 * 512);
  pack_w_kernel<<<grd(128 * 512), 256, 0, stream>>>(W_mu, WpML, 512, 16, 0,
                                                    128 * 512);
  pack_w_kernel<<<grd(128 * 512), 256, 0, stream>>>(W_ls, WpML, 512, 16, 128,
                                                    128 * 512);
  pack_w_kernel<<<grd(1024 * 512), 256, 0, stream>>>(W_wr, WpWR, 512, 64, 0,
                                                     1024 * 512);
  bias_concat_kernel<<<1, 256, 0, stream>>>(b_mu, b_ls, bml);
  img_pack_kernel<<<grd(4194304), 256, 0, stream>>>(x, img_p);
  zero_f32_kernel<<<grd(4194304), 256, 0, stream>>>(reconsts, 4194304);
  zero_f32_kernel<<<grd(131072), 256, 0, stream>>>(h_enc, 131072);
  zero_f32_kernel<<<grd(131072), 256, 0, stream>>>(h_dec, 131072);
  zero_bf16_kernel<<<grd(131072), 256, 0, stream>>>(h_enc_bf, 131072);
  zero_bf16_kernel<<<grd(131072), 256, 0, stream>>>(h_dec_bf, 131072);
  zero_bf16_kernel<<<grd(655360), 256, 0, stream>>>(enc_in, 655360);

  dim3 gBig(1536 / TN, 256 / TM);
  dim3 gML(256 / TN, 256 / TM);
  dim3 gWR(1024 / TN, 256 / TM);

  for (int t = 0; t < 16; ++t) {
    xhat_pack_kernel<<<grd(4194304), 256, 0, stream>>>(x, reconsts, xhat_p);
    fb_kernel<<<256, 128, 0, stream>>>(h_dec, W_ra, b_ra, gamma_r, Fx_p1, Fx_p2,
                                       Fy_n, Fy_t);
    read_attn_kernel<<<256, 256, 0, stream>>>(Fy_n, Fx_p1, img_p, xhat_p,
                                              gamma_r, enc_in);
    gemm_bf16_wmma<<<gBig, 256, 0, stream>>>(enc_in, WpIE, gi, b_ih_enc, 256,
                                             1536, 2560);
    gemm_bf16_wmma<<<gBig, 256, 0, stream>>>(h_enc_bf, WpHE, gh, b_hh_enc, 256,
                                             1536, 512);
    gru_fuse_kernel<<<512, 256, 0, stream>>>(gi, gh, h_enc, h_enc_bf, nullptr);
    gemm_bf16_wmma<<<gML, 256, 0, stream>>>(h_enc_bf, WpML, gml, bml, 256, 256,
                                            512);
    zfuse_kernel<<<128, 256, 0, stream>>>(gml, eps + (size_t)t * 32768, zb);
    gemm_bf16_wmma<<<gBig, 256, 0, stream>>>(zb, WpID, gi, b_ih_dec, 256, 1536,
                                             128);
    gemm_bf16_wmma<<<gBig, 256, 0, stream>>>(h_dec_bf, WpHD, gh, b_hh_dec, 256,
                                             1536, 512);
    gru_fuse_kernel<<<512, 256, 0, stream>>>(gi, gh, h_dec, h_dec_bf, enc_in);
    fb_kernel<<<256, 128, 0, stream>>>(h_dec, W_wa, b_wa, gamma_w, Fx_p1, Fx_p2,
                                       Fy_n, Fy_t);
    gemm_bf16_wmma<<<gWR, 256, 0, stream>>>(h_dec_bf, WpWR, gi, b_wr, 256, 1024,
                                            512);
    wpack_kernel<<<grd(262144), 256, 0, stream>>>(gi, wbp);
    write_attn_kernel<<<256, 256, 0, stream>>>(Fy_t, Fx_p2, wbp, gamma_w,
                                               reconsts);
  }
  sigmoid_out_kernel<<<grd(4194304), 256, 0, stream>>>(reconsts, (float*)d_out,
                                                       4194304);
  (void)in_sizes; (void)n_in; (void)out_size; (void)ws_size;
}
